// GNN_Example_27023934226651
// MI455X (gfx1250) — matrix operand order repaired and checked by reading, not compiled, not run on hardware
//
#include <hip/hip_runtime.h>
#include <hip/hip_bf16.h>

typedef __attribute__((ext_vector_type(16))) __bf16 v16bf;
typedef __attribute__((ext_vector_type(8)))  float  v8f;

// ---------------- gcn_norm ----------------

__global__ void tag_deg_kernel(const int* __restrict__ dst, const float* __restrict__ w,
                               float* __restrict__ deg, int E) {
    int e = blockIdx.x * blockDim.x + threadIdx.x;
    if (e >= E) return;
    __hip_atomic_fetch_add(&deg[dst[e]], w[e], __ATOMIC_RELAXED, __HIP_MEMORY_SCOPE_AGENT);
}

__global__ void tag_rsqrt_kernel(float* __restrict__ deg, int N) {
    int i = blockIdx.x * blockDim.x + threadIdx.x;
    if (i >= N) return;
    float d = deg[i];
    deg[i] = (d > 0.0f) ? rsqrtf(d) : 0.0f;   // deg buffer becomes dinv in place
}

__global__ void tag_norm_kernel(const int* __restrict__ src, const int* __restrict__ dst,
                                const float* __restrict__ w, const float* __restrict__ dinv,
                                float* __restrict__ nrm, int E) {
    int e = blockIdx.x * blockDim.x + threadIdx.x;
    if (e >= E) return;
    nrm[e] = dinv[src[e]] * w[e] * dinv[dst[e]];
}

// ---------------- weight prep: fp32 [4][128][128] (hop,in,out) -> bf16 [128][512] (out, K=hop*128+in) ----------------

__global__ void tag_prep_w_kernel(const float* __restrict__ W0, const float* __restrict__ W1,
                                  const float* __restrict__ W2, __bf16* __restrict__ Wt) {
    int idx = blockIdx.x * blockDim.x + threadIdx.x;   // 3 * 128 * 512 = 196608 threads
    if (idx >= 3 * 128 * 512) return;
    int l = idx >> 16;            // / 65536
    int r = idx & 65535;
    int n = r >> 9;               // out column 0..127
    int k = r & 511;              // fused K
    int hop = k >> 7;
    int i   = k & 127;            // input index
    const float* W = (l == 0) ? W0 : (l == 1) ? W1 : W2;
    Wt[idx] = (__bf16)W[hop * 16384 + i * 128 + n];
}

// ---------------- SpMM hop: hout[dst] += norm * hin[src], 128 feats, one wave per edge ----------------

__global__ void tag_spmm_kernel(const int* __restrict__ src, const int* __restrict__ dst,
                                const float* __restrict__ nrm, const float* __restrict__ hin,
                                float* __restrict__ hout, int E) {
    int e = blockIdx.x * 8 + (threadIdx.x >> 5);
    if (e >= E) return;
    int lane = threadIdx.x & 31;
    int s = src[e];
    int d = dst[e];
    float nv = nrm[e];
    float4 v = *(const float4*)(hin + (size_t)s * 128 + lane * 4);
    float* o = hout + (size_t)d * 128 + lane * 4;
    __hip_atomic_fetch_add(o + 0, nv * v.x, __ATOMIC_RELAXED, __HIP_MEMORY_SCOPE_AGENT);
    __hip_atomic_fetch_add(o + 1, nv * v.y, __ATOMIC_RELAXED, __HIP_MEMORY_SCOPE_AGENT);
    __hip_atomic_fetch_add(o + 2, nv * v.z, __ATOMIC_RELAXED, __HIP_MEMORY_SCOPE_AGENT);
    __hip_atomic_fetch_add(o + 3, nv * v.w, __ATOMIC_RELAXED, __HIP_MEMORY_SCOPE_AGENT);
}

// ---------------- fused TAGConv GEMM: C = [A0|A1|A2|A3] @ Wt^T (K=512) + bias, optional PReLU ----------------
// One wave = one 16x16 C tile; block = 8 waves = the 8 n-tiles of one m-tile.

__global__ __launch_bounds__(256) void tag_gemm_kernel(
    const float* __restrict__ A0, const float* __restrict__ A1,
    const float* __restrict__ A2, const float* __restrict__ A3,
    const __bf16* __restrict__ Wt,           // [128][512], K contiguous
    const float* __restrict__ bias,          // [128]
    const float* __restrict__ slope,         // [1] or nullptr -> no PReLU
    float* __restrict__ C, int N) {

    int wave = threadIdx.x >> 5;             // 0..7 -> n-tile
    int mt   = blockIdx.x;                   // m-tile
    if (mt * 16 >= N) return;
    int lane = threadIdx.x & 31;
    int ml   = lane & 15;
    int half = lane >> 4;

    int row = mt * 16 + ml;
    if (row >= N) row = N - 1;               // clamp loads; stores predicated below

    const float* Ap[4] = {A0, A1, A2, A3};
    v8f acc = {};

#pragma unroll
    for (int ks = 0; ks < 16; ++ks) {        // K = 512, 32 per WMMA
        const float* A = Ap[ks >> 2];
        int kk0 = (ks & 3) * 32;

        // A fragment (16-bit 16x32 layout): lane holds row=ml,
        // elems 0..15 -> K = kk0 + 16*half + i (matches B)
        const float* ar = A + (size_t)row * 128 + kk0 + half * 16;
        float4 a0 = *(const float4*)(ar + 0);
        float4 a1 = *(const float4*)(ar + 4);
        float4 a2 = *(const float4*)(ar + 8);
        float4 a3 = *(const float4*)(ar + 12);
        v16bf av;
        av[0]  = (__bf16)a0.x; av[1]  = (__bf16)a0.y; av[2]  = (__bf16)a0.z; av[3]  = (__bf16)a0.w;
        av[4]  = (__bf16)a1.x; av[5]  = (__bf16)a1.y; av[6]  = (__bf16)a1.z; av[7]  = (__bf16)a1.w;
        av[8]  = (__bf16)a2.x; av[9]  = (__bf16)a2.y; av[10] = (__bf16)a2.z; av[11] = (__bf16)a2.w;
        av[12] = (__bf16)a3.x; av[13] = (__bf16)a3.y; av[14] = (__bf16)a3.z; av[15] = (__bf16)a3.w;

        // B fragment (32x16): lane holds col n = ml, K = 16*half + i, contiguous in Wt
        const __bf16* bp = Wt + (size_t)(wave * 16 + ml) * 512 + ks * 32 + half * 16;
        v16bf bv = *(const v16bf*)bp;

        acc = __builtin_amdgcn_wmma_f32_16x16x32_bf16(false, av, false, bv,
                                                      (short)0, acc, false, false);
    }

    // Epilogue: bias + optional PReLU. C layout: VGPR r -> M = r + 8*half, N = ml.
    int col = wave * 16 + ml;
    float b = bias[col];
    bool doprelu = (slope != nullptr);
    float sl = doprelu ? slope[0] : 0.0f;
#pragma unroll
    for (int r = 0; r < 8; ++r) {
        int m = mt * 16 + r + half * 8;
        if (m < N) {
            float v = acc[r] + b;
            if (doprelu && v < 0.0f) v *= sl;
            C[(size_t)m * 128 + col] = v;
        }
    }
}

// ---------------- host ----------------

extern "C" void kernel_launch(void* const* d_in, const int* in_sizes, int n_in,
                              void* d_out, int out_size, void* d_ws, size_t ws_size,
                              hipStream_t stream) {
    const float* x  = (const float*)d_in[0];
    const int*   ei = (const int*)d_in[1];
    const float* ew = (const float*)d_in[2];
    const float* W0 = (const float*)d_in[3];
    const float* b0 = (const float*)d_in[4];
    const float* W1 = (const float*)d_in[5];
    const float* b1 = (const float*)d_in[6];
    const float* W2 = (const float*)d_in[7];
    const float* b2 = (const float*)d_in[8];
    const float* p0 = (const float*)d_in[9];
    const float* p1 = (const float*)d_in[10];

    int N = in_sizes[0] / 128;
    int E = in_sizes[1] / 2;
    const int* src = ei;
    const int* dst = ei + E;

    // bump allocator on workspace, 256B aligned
    char* wp = (char*)d_ws;
    auto alloc = [&](size_t bytes) -> void* {
        void* r = (void*)wp;
        wp += (bytes + 255) & ~(size_t)255;
        return r;
    };
    float*  dinv = (float*)alloc((size_t)N * 4);           // deg -> dinv in place
    float*  nrm  = (float*)alloc((size_t)E * 4);
    __bf16* Wt   = (__bf16*)alloc((size_t)3 * 128 * 512 * 2);
    size_t  hb   = (size_t)N * 128 * 4;
    float*  h1   = (float*)alloc(hb);
    float*  h2   = (float*)alloc(hb);
    float*  h3   = (float*)alloc(hb);
    float*  o0   = (float*)alloc(hb);
    float*  o1   = (float*)alloc(hb);

    // gcn_norm
    hipMemsetAsync(dinv, 0, (size_t)N * 4, stream);
    tag_deg_kernel<<<(E + 255) / 256, 256, 0, stream>>>(dst, ew, dinv, E);
    tag_rsqrt_kernel<<<(N + 255) / 256, 256, 0, stream>>>(dinv, N);
    tag_norm_kernel<<<(E + 255) / 256, 256, 0, stream>>>(src, dst, ew, dinv, nrm, E);

    // weights -> bf16 transposed once
    tag_prep_w_kernel<<<(3 * 128 * 512 + 255) / 256, 256, 0, stream>>>(W0, W1, W2, Wt);

    const float* biases[3] = {b0, b1, b2};
    const float* slopes[3] = {p0, p1, nullptr};
    float*       outs[3]   = {o0, o1, (float*)d_out};

    const float* Ain = x;
    int spmm_blocks = (E + 7) / 8;
    int gemm_blocks = (N + 15) / 16;

    for (int l = 0; l < 3; ++l) {
        hipMemsetAsync(h1, 0, hb, stream);
        tag_spmm_kernel<<<spmm_blocks, 256, 0, stream>>>(src, dst, nrm, Ain, h1, E);
        hipMemsetAsync(h2, 0, hb, stream);
        tag_spmm_kernel<<<spmm_blocks, 256, 0, stream>>>(src, dst, nrm, h1, h2, E);
        hipMemsetAsync(h3, 0, hb, stream);
        tag_spmm_kernel<<<spmm_blocks, 256, 0, stream>>>(src, dst, nrm, h2, h3, E);

        tag_gemm_kernel<<<gemm_blocks, 256, 0, stream>>>(
            Ain, h1, h2, h3, Wt + (size_t)l * 128 * 512,
            biases[l], slopes[l], outs[l], N);

        Ain = outs[l];
    }
}
